// MultiHeadCausalAttention_34119220199744
// MI455X (gfx1250) — matrix operand order repaired and checked by reading, not compiled, not run on hardware
//
#include <hip/hip_runtime.h>

typedef unsigned int   u32;
typedef unsigned short u16;
typedef __attribute__((ext_vector_type(16))) __bf16 v16bf;
typedef __attribute__((ext_vector_type(8)))  float  v8f;
typedef __attribute__((ext_vector_type(4)))  u32    u32x4;

union Frag {
  v16bf v;
  u32x4 q[2];
};

// ---- fp32 -> bf16 (round to nearest even) ----
__device__ __forceinline__ u16 f2bf(float f) {
  u32 u = __float_as_uint(f);
  u += 0x7FFFu + ((u >> 16) & 1u);
  return (u16)(u >> 16);
}

// async 16B global->LDS copy (ASYNCcnt path); lds = 32-bit LDS offset
__device__ __forceinline__ void async_cp16(u32 lds, const u16* g) {
  asm volatile("global_load_async_to_lds_b128 %0, %1, off"
               :: "v"(lds), "v"(g) : "memory");
}
__device__ __forceinline__ void wait_async0() {
  asm volatile("s_wait_asynccnt 0" ::: "memory");
}
__device__ __forceinline__ u32 lds_off(const void* p) {
  return (u32)(size_t)p;  // addrspace(3) offset lives in the low 32 bits
}

// A-layout fragment: 16(M) x 32(K) bf16 tile, src row-major, leading dim ld (elements).
// ISA 16-bit A layout: lanes 0-15 M=lane K{0..7,16..23}; lanes 16-31 M=lane-16 K{8..15,24..31}
__device__ __forceinline__ v16bf load_frag_a(const u16* src, int ld, int lane) {
  Frag f;
  const int row  = lane & 15;
  const int koff = (lane >> 4) * 8;
  const u16* p = src + row * ld + koff;
  f.q[0] = *(const u32x4*)(p);        // K koff..koff+7   -> elems 0..7  (VGPR 0..3)
  f.q[1] = *(const u32x4*)(p + 16);   // K koff+16..+23   -> elems 8..15 (VGPR 4..7)
  return f.v;
}

// B-layout fragment from "Bt" storage bt[n*ld + k] (k contiguous per column n).
// ISA 16-bit B layout: lanes 0-15 N=lane K{0..7,16..23}; lanes 16-31 N=lane-16 K{8..15,24..31}
__device__ __forceinline__ v16bf load_frag_b(const u16* bt, int ld, int lane) {
  Frag f;
  const int n    = lane & 15;
  const int koff = (lane >> 4) * 8;
  const u16* p = bt + n * ld + koff;
  f.q[0] = *(const u32x4*)(p);
  f.q[1] = *(const u32x4*)(p + 16);
  return f.v;
}

__device__ __forceinline__ v8f wmma_bf16(v16bf a, v16bf b, v8f c) {
  return __builtin_amdgcn_wmma_f32_16x16x32_bf16(false, a, false, b, (short)0, c,
                                                 false, false);
}

// ---------------- fp32 -> bf16 convert ----------------
__global__ __launch_bounds__(256) void k_cvt(const float* __restrict__ s,
                                             u16* __restrict__ d, int n) {
  int i = blockIdx.x * 256 + threadIdx.x;
  if (i < n) d[i] = f2bf(s[i]);
}

// ---------------- tiled bf16 GEMM: C[M,N] = A[M,K] * B[K,N] ----------------
// block tile 128x128, 8 waves, each wave 32x64 (2x4 tiles of 16x16), K step 64
template <bool OUT_BF16>
__global__ __launch_bounds__(256) void k_gemm(const u16* __restrict__ A,
                                              const u16* __restrict__ B,
                                              void* __restrict__ Cout,
                                              int M, int N, int K) {
  __shared__ u16 As[128 * 64];   // [m][k] row-major      (16 KB, async-staged)
  __shared__ u16 Bts[128 * 64];  // [n][k] transposed     (16 KB, manual transpose)

  const int tid  = threadIdx.x;
  const int lane = tid & 31;
  const int wave = tid >> 5;
  const int wm   = wave >> 1;    // 0..3  -> 32-row strip
  const int wn   = wave & 1;     // 0..1  -> 64-col strip
  const int m0   = blockIdx.x * 128;
  const int n0   = blockIdx.y * 128;

  v8f acc[2][4];
#pragma unroll
  for (int i = 0; i < 2; ++i)
#pragma unroll
    for (int j = 0; j < 4; ++j) acc[i][j] = {};

  for (int k0 = 0; k0 < K; k0 += 64) {
    // stage A tile: 128x64, 1024 16B chunks via async global->LDS
#pragma unroll
    for (int rep = 0; rep < 4; ++rep) {
      const int c   = rep * 256 + tid;
      const int row = c >> 3, off = (c & 7) * 8;
      async_cp16(lds_off(&As[row * 64 + off]),
                 A + (size_t)(m0 + row) * K + k0 + off);
    }
    // stage B tile transposed into Bts[n][k]: 64 k-rows x 128 n
#pragma unroll
    for (int rep = 0; rep < 2; ++rep) {
      const int lid = rep * 256 + tid;
      const int k = lid >> 3, nc = (lid & 7) * 16;
      const u16* g = B + (size_t)(k0 + k) * N + n0 + nc;
      union { u32x4 q; u16 e[8]; } t0, t1;
      t0.q = *(const u32x4*)g;
      t1.q = *(const u32x4*)(g + 8);
#pragma unroll
      for (int j = 0; j < 8; ++j) {
        Bts[(nc + j) * 64 + k]     = t0.e[j];
        Bts[(nc + 8 + j) * 64 + k] = t1.e[j];
      }
    }
    wait_async0();        // our async LDS writes landed
    __syncthreads();      // everyone's staging visible

#pragma unroll
    for (int ks = 0; ks < 2; ++ks) {
      v16bf af0 = load_frag_a(&As[(wm * 32) * 64 + ks * 32], 64, lane);
      v16bf af1 = load_frag_a(&As[(wm * 32 + 16) * 64 + ks * 32], 64, lane);
#pragma unroll
      for (int j = 0; j < 4; ++j) {
        v16bf bf = load_frag_b(&Bts[(wn * 64 + j * 16) * 64 + ks * 32], 64, lane);
        acc[0][j] = wmma_bf16(af0, bf, acc[0][j]);
        acc[1][j] = wmma_bf16(af1, bf, acc[1][j]);
      }
    }
    __syncthreads();
  }

  // epilogue: C/D layout -> row = r + 8*(lane>>4), col = lane&15
  const int hh = lane >> 4, col = lane & 15;
#pragma unroll
  for (int i = 0; i < 2; ++i)
#pragma unroll
    for (int j = 0; j < 4; ++j)
#pragma unroll
      for (int r = 0; r < 8; ++r) {
        const int row = m0 + wm * 32 + i * 16 + r + 8 * hh;
        const int cc  = n0 + wn * 64 + j * 16 + col;
        const float v = acc[i][j][r];
        if (OUT_BF16) ((u16*)Cout)[(size_t)row * N + cc]   = f2bf(v);
        else          ((float*)Cout)[(size_t)row * N + cc] = v;
      }
}

// ---------------- flash attention ----------------
// grid (T/64, H, B), block 128 (4 waves); each wave owns 16 q rows of the 64-row tile
__global__ __launch_bounds__(128) void k_attn(const u16* __restrict__ qkv,
                                              u16* __restrict__ attn) {
  constexpr int T = 2048, C = 1024, Dh = 64;
  const int qb = blockIdx.x, h = blockIdx.y, b = blockIdx.z;
  const int tid = threadIdx.x, lane = tid & 31, w = tid >> 5;
  const int hh = lane >> 4, col = lane & 15;
  const int qt0 = qb * 64;

  __shared__ u16 Kt[64 * 64];      // [krow][d]  (row-major K == Bt layout for Q*K^T)
  __shared__ u16 Vt[64 * 64];      // [d][krow]  (transposed V == Bt layout for P*V)
  __shared__ u16 Ps[4][16 * 64];   // per-wave P scratch, [qrow][kcol]

  const size_t ld = (size_t)3 * C;
  const u16* qbase = qkv + (size_t)b * T * ld + h * Dh;
  const u16* kbase = qbase + C;
  const u16* vbase = qbase + 2 * C;

  // Q fragments (A layout) loaded once, straight from global
  v16bf qf[2];
  {
    const int qrow = qt0 + w * 16 + col;
    const u16* qp = qbase + (size_t)qrow * ld;
#pragma unroll
    for (int c2 = 0; c2 < 2; ++c2) {
      Frag f;
      const u16* p = qp + c2 * 32 + hh * 8;
      f.q[0] = *(const u32x4*)(p);
      f.q[1] = *(const u32x4*)(p + 16);
      qf[c2] = f.v;
    }
  }

  v8f oacc[4];
#pragma unroll
  for (int j = 0; j < 4; ++j) oacc[j] = {};
  float mrow[8], lrow[8];
#pragma unroll
  for (int r = 0; r < 8; ++r) { mrow[r] = -3.0e38f; lrow[r] = 0.f; }

  for (int kt = 0; kt <= qb; ++kt) {
    // ---- K tile: straight copy -> async global->LDS (512 16B chunks) ----
#pragma unroll
    for (int rep = 0; rep < 4; ++rep) {
      const int cid  = rep * 128 + tid;
      const int krow = cid >> 3, off = (cid & 7) * 8;
      async_cp16(lds_off(&Kt[krow * 64 + off]),
                 kbase + (size_t)(kt * 64 + krow) * ld + off);
    }
    // ---- V tile: transposed into Vt[d][krow] (manual) ----
#pragma unroll
    for (int rep = 0; rep < 2; ++rep) {
      const int cid = rep * 128 + tid;
      const int krow = cid >> 2, dc = (cid & 3) * 16;
      const u16* gv = vbase + (size_t)(kt * 64 + krow) * ld + dc;
      union { u32x4 q; u16 e[8]; } t0, t1;
      t0.q = *(const u32x4*)gv;
      t1.q = *(const u32x4*)(gv + 8);
#pragma unroll
      for (int j = 0; j < 8; ++j) {
        Vt[(dc + j) * 64 + krow]     = t0.e[j];
        Vt[(dc + 8 + j) * 64 + krow] = t1.e[j];
      }
    }
    if (kt + 1 <= qb)  // warm L2 for next K tile
      __builtin_prefetch(kbase + (size_t)((kt + 1) * 64) * ld, 0, 1);
    wait_async0();
    __syncthreads();

    // ---- S = (Q * K^T) * 1/sqrt(Dh), 4 column tiles of 16 ----
    v8f s[4];
#pragma unroll
    for (int kc = 0; kc < 4; ++kc) {
      v8f a = {};
#pragma unroll
      for (int c2 = 0; c2 < 2; ++c2) {
        v16bf kf = load_frag_b(&Kt[(kc * 16) * 64 + c2 * 32], 64, lane);
        a = wmma_bf16(qf[c2], kf, a);
      }
      a = a * 0.125f;
      if (kt == qb) {  // causal mask only on diagonal tile
        const int cg = kt * 64 + kc * 16 + col;
#pragma unroll
        for (int r = 0; r < 8; ++r) {
          const int qg = qt0 + w * 16 + r + 8 * hh;
          if (cg > qg) a[r] = -3.0e38f;
        }
      }
      s[kc] = a;
    }

    // ---- online softmax (rows live in VGPR index + lane-half; cols across 16 lanes) ----
#pragma unroll
    for (int r = 0; r < 8; ++r) {
      float vmax = fmaxf(fmaxf(s[0][r], s[1][r]), fmaxf(s[2][r], s[3][r]));
#pragma unroll
      for (int x = 1; x < 16; x <<= 1) vmax = fmaxf(vmax, __shfl_xor(vmax, x, 32));
      const float mnew = fmaxf(mrow[r], vmax);
      const float sc = __expf(mrow[r] - mnew);
      mrow[r] = mnew;
      float rs = 0.f;
#pragma unroll
      for (int kc = 0; kc < 4; ++kc) {
        const float p = __expf(s[kc][r] - mnew);
        rs += p;
        Ps[w][(r + 8 * hh) * 64 + kc * 16 + col] = f2bf(p);
      }
#pragma unroll
      for (int x = 1; x < 16; x <<= 1) rs += __shfl_xor(rs, x, 32);
      lrow[r] = lrow[r] * sc + rs;
#pragma unroll
      for (int j = 0; j < 4; ++j) oacc[j][r] *= sc;
    }
    // per-wave LDS RAW: P stores must land before A-fragment reloads
    asm volatile("s_wait_dscnt 0" ::: "memory");

    // ---- O += P * V ----
#pragma unroll
    for (int c2 = 0; c2 < 2; ++c2) {
      v16bf pf = load_frag_a(&Ps[w][c2 * 32], 64, lane);
#pragma unroll
      for (int j = 0; j < 4; ++j) {
        v16bf vf = load_frag_b(&Vt[(j * 16) * 64 + c2 * 32], 64, lane);
        oacc[j] = wmma_bf16(pf, vf, oacc[j]);
      }
    }
    __syncthreads();  // before next iteration overwrites Kt/Vt
  }

  // ---- normalize and store bf16 head output into [B,T,C] ----
#pragma unroll
  for (int j = 0; j < 4; ++j)
#pragma unroll
    for (int r = 0; r < 8; ++r) {
      const int qrow = qt0 + w * 16 + r + 8 * hh;
      const float v = oacc[j][r] / lrow[r];
      attn[((size_t)b * T + qrow) * C + h * Dh + j * 16 + col] = f2bf(v);
    }
}

// ---------------- host launcher ----------------
extern "C" void kernel_launch(void* const* d_in, const int* in_sizes, int n_in,
                              void* d_out, int out_size, void* d_ws, size_t ws_size,
                              hipStream_t stream) {
  (void)in_sizes; (void)n_in; (void)out_size; (void)ws_size;
  const float* x    = (const float*)d_in[0];
  const float* Wqkv = (const float*)d_in[1];
  const float* Wout = (const float*)d_in[2];
  float* out = (float*)d_out;

  constexpr int B = 4, T = 2048, C = 1024;
  constexpr size_t nX   = (size_t)B * T * C;       // 8388608
  constexpr size_t nWq  = (size_t)C * 3 * C;       // 3145728
  constexpr size_t nWo  = (size_t)C * C;           // 1048576
  constexpr size_t nQKV = (size_t)B * T * 3 * C;   // 25165824

  u16* x_bf   = (u16*)d_ws;
  u16* wq_bf  = x_bf + nX;
  u16* wo_bf  = wq_bf + nWq;
  u16* qkv_bf = wo_bf + nWo;
  u16* at_bf  = qkv_bf + nQKV;   // total ws use: ~92.3 MB

  k_cvt<<<(unsigned)(nX / 256),  256, 0, stream>>>(x,    x_bf,  (int)nX);
  k_cvt<<<(unsigned)(nWq / 256), 256, 0, stream>>>(Wqkv, wq_bf, (int)nWq);
  k_cvt<<<(unsigned)(nWo / 256), 256, 0, stream>>>(Wout, wo_bf, (int)nWo);

  // qkv = x @ W_qkv : M=8192 N=3072 K=1024
  k_gemm<true><<<dim3(64, 24), 256, 0, stream>>>(x_bf, wq_bf, qkv_bf, 8192, 3072, 1024);

  // flash attention per (qtile, head, batch)
  k_attn<<<dim3(T / 64, 16, B), 128, 0, stream>>>(qkv_bf, at_bf);

  // out = attn @ W_out : M=8192 N=1024 K=1024
  k_gemm<false><<<dim3(64, 8), 256, 0, stream>>>(at_bf, wo_bf, out, 8192, 1024, 1024);
}
